// MultiModalFusionGNN_58909771432759
// MI455X (gfx1250) — compile-verified, matrix operand order checked by reading, not run on hardware
//
#include <hip/hip_runtime.h>

#define B_  2
#define N_  2048
#define H_  64
#define NH_ 4
#define HD_ 16
#define E_  32768
#define EMB (B_*N_*H_)

typedef __attribute__((ext_vector_type(16))) _Float16 v16h;
typedef __attribute__((ext_vector_type(8)))  float    v8f;

// ---------------- block-of-64 sum reduction (for LayerNorm) ----------------
__device__ __forceinline__ float blk64_sum(float v, float* red, int j) {
  red[j] = v;               __syncthreads();
  if (j < 32) red[j] += red[j + 32]; __syncthreads();
  if (j < 16) red[j] += red[j + 16]; __syncthreads();
  if (j <  8) red[j] += red[j +  8]; __syncthreads();
  if (j <  4) red[j] += red[j +  4]; __syncthreads();
  if (j <  2) red[j] += red[j +  2]; __syncthreads();
  if (j <  1) red[j] += red[j +  1]; __syncthreads();
  float r = red[0];         __syncthreads();
  return r;
}

// ---------------- encoder: Lin->ReLU->LN, Lin->ReLU->LN, +memb -------------
__global__ __launch_bounds__(64) void k_encode(
    const float* __restrict__ x, int D,
    const float* __restrict__ w1, const float* __restrict__ b1,
    const float* __restrict__ g1, const float* __restrict__ be1,
    const float* __restrict__ w2, const float* __restrict__ b2,
    const float* __restrict__ g2, const float* __restrict__ be2,
    const float* __restrict__ memb, float* __restrict__ out)
{
  int node = blockIdx.x, j = threadIdx.x;
  __shared__ float xs[8];
  __shared__ float h[64];
  __shared__ float red[64];
  if (j < D) xs[j] = x[(size_t)node * D + j];
  __syncthreads();
  float a = b1[j];
  for (int d = 0; d < D; ++d) a += xs[d] * w1[d * H_ + j];
  a = fmaxf(a, 0.f);
  float mean = blk64_sum(a, red, j) * (1.f / 64.f);
  float dv = a - mean;
  float var = blk64_sum(dv * dv, red, j) * (1.f / 64.f);
  h[j] = dv * rsqrtf(var + 1e-5f) * g1[j] + be1[j];
  __syncthreads();
  float a2 = b2[j];
  for (int k = 0; k < H_; ++k) a2 += h[k] * w2[k * H_ + j];
  a2 = fmaxf(a2, 0.f);
  float mean2 = blk64_sum(a2, red, j) * (1.f / 64.f);
  float dv2 = a2 - mean2;
  float var2 = blk64_sum(dv2 * dv2, red, j) * (1.f / 64.f);
  out[(size_t)node * H_ + j] = dv2 * rsqrtf(var2 + 1e-5f) * g2[j] + be2[j] + memb[j];
}

// ---------------- edge scatter-add (neighbor aggregation) ------------------
__global__ __launch_bounds__(64) void k_scatter(
    const float* __restrict__ emb, const int* __restrict__ src,
    const int* __restrict__ dst, float* __restrict__ nb)
{
  int e = blockIdx.x, b = blockIdx.y, f = threadIdx.x;
  int s = src[e], d = dst[e];
  atomicAdd(&nb[((size_t)b * N_ + d) * H_ + f], emb[((size_t)b * N_ + s) * H_ + f]);
}

// ---------------- impute MLP over [neighbor || emb] ------------------------
__global__ __launch_bounds__(64) void k_impute(
    const float* __restrict__ nb, const float* __restrict__ emb,
    const int* __restrict__ mask,
    const float* __restrict__ w1, const float* __restrict__ b1,
    const float* __restrict__ w2, const float* __restrict__ b2,
    float* __restrict__ out)
{
  int node = blockIdx.x, j = threadIdx.x;
  __shared__ float ctx[128];
  __shared__ float h[64];
  ctx[j]      = nb[(size_t)node * H_ + j];
  ctx[64 + j] = emb[(size_t)node * H_ + j];
  __syncthreads();
  float a = b1[j];
  for (int k = 0; k < 128; ++k) a += ctx[k] * w1[k * H_ + j];
  h[j] = fmaxf(a, 0.f);
  __syncthreads();
  float o = b2[j];
  for (int k = 0; k < 64; ++k) o += h[k] * w2[k * H_ + j];
  // missing = 1 - mask -> use imputed when mask == 0
  out[(size_t)node * H_ + j] = (mask[node] == 0) ? o : ctx[64 + j];
}

// ---------------- Q/K/V projections, stored f16 ----------------------------
__global__ __launch_bounds__(64) void k_qkv(
    const float* __restrict__ qm, const float* __restrict__ km, const float* __restrict__ vm,
    const float* __restrict__ wq, const float* __restrict__ bq,
    const float* __restrict__ wk, const float* __restrict__ bk,
    const float* __restrict__ wv, const float* __restrict__ bv,
    _Float16* __restrict__ Qh, _Float16* __restrict__ Kh, _Float16* __restrict__ Vh)
{
  int node = blockIdx.x, j = threadIdx.x;
  __shared__ float qs[64], ks[64], vs[64];
  qs[j] = qm[(size_t)node * H_ + j];
  ks[j] = km[(size_t)node * H_ + j];
  vs[j] = vm[(size_t)node * H_ + j];
  __syncthreads();
  float q = bq[j], k = bk[j], v = bv[j];
  for (int t = 0; t < 64; ++t) {
    q += qs[t] * wq[t * H_ + j];
    k += ks[t] * wk[t * H_ + j];
    v += vs[t] * wv[t * H_ + j];
  }
  Qh[(size_t)node * H_ + j] = (_Float16)q;
  Kh[(size_t)node * H_ + j] = (_Float16)k;
  Vh[(size_t)node * H_ + j] = (_Float16)v;
}

// ---------------- flash-attention core: one wave per (b, head, q-tile) -----
// Scores via v_wmma_f32_16x16x32_f16 (HD=16 zero-padded to K=32); streaming
// online softmax over 32-key chunks; P*V accumulated in the f32 C fragment.
// V chunk is staged into LDS with two wide global loads per lane, then the
// transposed B-fragment gather happens as conflict-free ds reads.
__global__ __launch_bounds__(32) void k_attn(
    const _Float16* __restrict__ Qm, const _Float16* __restrict__ Km,
    const _Float16* __restrict__ Vm, float* __restrict__ att)
{
  const int qt = blockIdx.x, head = blockIdx.y, b = blockIdx.z;
  const int lane = threadIdx.x;
  const int hl   = lane >> 4;    // which half of the wave
  const int c16  = lane & 15;
  const int qbase = qt * 16;

  __shared__ float Stile[16][32];
  __shared__ __align__(16) _Float16 Vtile[32][16];  // [key-in-chunk][hd]
  __shared__ float alphaA[16];
  __shared__ float lA[16];

  // A-layout Q fragment: lane row = c16; halves 0..7 hold hd (hl*8 .. hl*8+7),
  // halves 8..15 are the K=16..31 zero padding.
  v16h qa;
  {
    const _Float16* qp = Qm + ((size_t)(b * N_) + qbase + c16) * H_ + head * HD_ + hl * 8;
    #pragma unroll
    for (int e = 0; e < 8; ++e) qa[e] = qp[e];
    #pragma unroll
    for (int e = 8; e < 16; ++e) qa[e] = (_Float16)0.f;
  }

  v8f acc, zero8;
  #pragma unroll
  for (int r = 0; r < 8; ++r) { acc[r] = 0.f; zero8[r] = 0.f; }
  float mrun = -1e30f, lrun = 0.f;

  for (int kb = 0; kb < N_ / 32; ++kb) {
    const int kbase = kb * 32;
    const _Float16* kp0 = Km + ((size_t)(b * N_) + kbase + c16) * H_ + head * HD_ + hl * 8;
    const _Float16* kp1 = kp0 + (size_t)16 * H_;
    if (kb + 1 < N_ / 32) {
      __builtin_prefetch(kp0 + (size_t)32 * H_, 0, 1);
      __builtin_prefetch(Vm + ((size_t)(b * N_) + kbase + 32 + lane) * H_ + head * HD_, 0, 1);
    }

    // cooperative V stage: lane loads key-row (kbase+lane)'s 16-half head
    // slice with two 16-byte loads -> LDS
    {
      const _Float16* vrow = Vm + ((size_t)(b * N_) + kbase + lane) * H_ + head * HD_;
      uint4 t0 = *(const uint4*)(vrow);
      uint4 t1 = *(const uint4*)(vrow + 8);
      *(uint4*)(&Vtile[lane][0]) = t0;
      *(uint4*)(&Vtile[lane][8]) = t1;
    }

    // B-layout K^T fragments: lane col = key index, elements = hd (padded)
    v16h kf0, kf1;
    #pragma unroll
    for (int e = 0; e < 8; ++e) { kf0[e] = kp0[e]; kf1[e] = kp1[e]; }
    #pragma unroll
    for (int e = 8; e < 16; ++e) { kf0[e] = (_Float16)0.f; kf1[e] = (_Float16)0.f; }

    v8f s0 = __builtin_amdgcn_wmma_f32_16x16x32_f16(false, qa, false, kf0, (short)0, zero8, false, false);
    v8f s1 = __builtin_amdgcn_wmma_f32_16x16x32_f16(false, qa, false, kf1, (short)0, zero8, false, false);

    // C layout -> LDS [qrow][key], scaled by 1/sqrt(HD)
    #pragma unroll
    for (int r = 0; r < 8; ++r) {
      int row = r + hl * 8;
      Stile[row][c16]      = s0[r] * 0.25f;
      Stile[row][16 + c16] = s1[r] * 0.25f;
    }
    __syncthreads();

    // online softmax bookkeeping: one lane per query row
    if (lane < 16) {
      float rmax = -1e30f;
      #pragma unroll
      for (int c = 0; c < 32; ++c) rmax = fmaxf(rmax, Stile[lane][c]);
      float nm = fmaxf(mrun, rmax);
      float al = __expf(mrun - nm);
      float rs = 0.f;
      #pragma unroll
      for (int c = 0; c < 32; ++c) {
        float p = __expf(Stile[lane][c] - nm);
        Stile[lane][c] = p;
        rs += p;
      }
      lrun = lrun * al + rs;
      mrun = nm;
      alphaA[lane] = al;
    }
    __syncthreads();

    // rescale accumulator rows by alpha
    #pragma unroll
    for (int r = 0; r < 8; ++r) acc[r] *= alphaA[r + hl * 8];

    // A-layout P fragment and B-layout V fragment, both gathered from LDS
    v16h pf, vf;
    #pragma unroll
    for (int e = 0; e < 16; ++e) {
      int kk = (e & 7) + ((e >> 3) * 16) + hl * 8;   // key offset 0..31
      pf[e] = (_Float16)Stile[c16][kk];
      vf[e] = Vtile[kk][c16];
    }
    acc = __builtin_amdgcn_wmma_f32_16x16x32_f16(false, pf, false, vf, (short)0, acc, false, false);
    __syncthreads();
  }

  if (lane < 16) lA[lane] = lrun;
  __syncthreads();
  #pragma unroll
  for (int r = 0; r < 8; ++r) {
    int row = r + hl * 8;
    att[((size_t)(b * N_) + qbase + row) * H_ + head * HD_ + c16] = acc[r] / lA[row];
  }
}

// ---------------- output projection + residual + LN, 0.5x accumulate -------
__global__ __launch_bounds__(64) void k_projln(
    const float* __restrict__ att, const float* __restrict__ qm,
    const float* __restrict__ wo, const float* __restrict__ bo,
    const float* __restrict__ g, const float* __restrict__ be,
    float* __restrict__ acc)
{
  int node = blockIdx.x, j = threadIdx.x;
  __shared__ float as[64];
  __shared__ float red[64];
  as[j] = att[(size_t)node * H_ + j];
  __syncthreads();
  float o = bo[j];
  for (int k = 0; k < 64; ++k) o += as[k] * wo[k * H_ + j];
  o += qm[(size_t)node * H_ + j];
  float mean = blk64_sum(o, red, j) * (1.f / 64.f);
  float dv = o - mean;
  float var = blk64_sum(dv * dv, red, j) * (1.f / 64.f);
  acc[(size_t)node * H_ + j] += 0.5f * (dv * rsqrtf(var + 1e-5f) * g[j] + be[j]);
}

// ---------------- modal-weighted fusion: Lin(192->64)->ReLU->LN ------------
__global__ __launch_bounds__(64) void k_fuse(
    const float* __restrict__ pfu, const float* __restrict__ ffu, const float* __restrict__ afu,
    const float* __restrict__ mi,
    const float* __restrict__ w, const float* __restrict__ bb,
    const float* __restrict__ g, const float* __restrict__ be,
    float* __restrict__ fused)
{
  int node = blockIdx.x, j = threadIdx.x;
  __shared__ float ctx[192];
  __shared__ float red[64];
  float m0 = mi[0], m1 = mi[1], m2 = mi[2];
  float mx = fmaxf(m0, fmaxf(m1, m2));
  float e0 = __expf(m0 - mx), e1 = __expf(m1 - mx), e2 = __expf(m2 - mx);
  float inv = 1.f / (e0 + e1 + e2);
  ctx[j]       = e0 * inv * pfu[(size_t)node * H_ + j];
  ctx[64 + j]  = e1 * inv * ffu[(size_t)node * H_ + j];
  ctx[128 + j] = e2 * inv * afu[(size_t)node * H_ + j];
  __syncthreads();
  float a = bb[j];
  for (int k = 0; k < 192; ++k) a += ctx[k] * w[k * H_ + j];
  a = fmaxf(a, 0.f);
  float mean = blk64_sum(a, red, j) * (1.f / 64.f);
  float dv = a - mean;
  float var = blk64_sum(dv * dv, red, j) * (1.f / 64.f);
  fused[(size_t)node * H_ + j] = dv * rsqrtf(var + 1e-5f) * g[j] + be[j];
}

// ---------------- node head: 64->32->1 -------------------------------------
__global__ __launch_bounds__(64) void k_nodehead(
    const float* __restrict__ fused,
    const float* __restrict__ w1, const float* __restrict__ b1,
    const float* __restrict__ w2, const float* __restrict__ b2,
    float* __restrict__ out)
{
  int node = blockIdx.x, j = threadIdx.x;
  __shared__ float fs[64];
  __shared__ float h1[32];
  fs[j] = fused[(size_t)node * H_ + j];
  __syncthreads();
  if (j < 32) {
    float a = b1[j];
    for (int k = 0; k < 64; ++k) a += fs[k] * w1[k * 32 + j];
    h1[j] = fmaxf(a, 0.f);
  }
  __syncthreads();
  if (j == 0) {
    float o = b2[0];
    for (int k = 0; k < 32; ++k) o += h1[k] * w2[k];
    out[node] = o;
  }
}

// ---------------- graph head: mean over N then 64->32->1 -------------------
__global__ __launch_bounds__(64) void k_graphhead(
    const float* __restrict__ fused,
    const float* __restrict__ w1, const float* __restrict__ b1,
    const float* __restrict__ w2, const float* __restrict__ b2,
    float* __restrict__ out)
{
  int b = blockIdx.x, j = threadIdx.x;
  __shared__ float g[64];
  __shared__ float h1[32];
  float s = 0.f;
  for (int n = 0; n < N_; ++n) s += fused[((size_t)b * N_ + n) * H_ + j];
  g[j] = s * (1.f / (float)N_);
  __syncthreads();
  if (j < 32) {
    float a = b1[j];
    for (int k = 0; k < 64; ++k) a += g[k] * w1[k * 32 + j];
    h1[j] = fmaxf(a, 0.f);
  }
  __syncthreads();
  if (j == 0) {
    float o = b2[0];
    for (int k = 0; k < 32; ++k) o += h1[k] * w2[k];
    out[b] = o;
  }
}

// ===========================================================================
extern "C" void kernel_launch(void* const* d_in, const int* in_sizes, int n_in,
                              void* d_out, int out_size, void* d_ws, size_t ws_size,
                              hipStream_t stream)
{
  (void)in_sizes; (void)n_in; (void)out_size; (void)ws_size;

  const float* pf = (const float*)d_in[0];
  const float* ff = (const float*)d_in[1];
  const float* af = (const float*)d_in[2];
  const int* pm = (const int*)d_in[3];
  const int* fm = (const int*)d_in[4];
  const int* am = (const int*)d_in[5];
  const int* ei = (const int*)d_in[6];
  // d_in[7] = edge_attr (unused by the reference math)
  auto P = [&](int i) { return (const float*)d_in[i]; };

  float* ws    = (float*)d_ws;
  float* pe    = ws + (size_t)0  * EMB;
  float* fe    = ws + (size_t)1  * EMB;
  float* ae    = ws + (size_t)2  * EMB;
  float* pei   = ws + (size_t)3  * EMB;
  float* fei   = ws + (size_t)4  * EMB;
  float* aei   = ws + (size_t)5  * EMB;
  float* nb    = ws + (size_t)6  * EMB;
  float* pfu   = ws + (size_t)7  * EMB;
  float* ffu   = ws + (size_t)8  * EMB;
  float* afu   = ws + (size_t)9  * EMB;
  float* fused = ws + (size_t)10 * EMB;
  float* attb  = ws + (size_t)11 * EMB;
  _Float16* Qh = (_Float16*)(ws + (size_t)12 * EMB);
  _Float16* Kh = Qh + (size_t)EMB;
  _Float16* Vh = Kh + (size_t)EMB;

  dim3 nodeGrid(B_ * N_);

  // encoders (param bases: enc_p=8, enc_f=17, enc_a=26)
  k_encode<<<nodeGrid, 64, 0, stream>>>(pf, 2, P(8),  P(9),  P(10), P(11), P(12), P(13), P(14), P(15), P(16), pe);
  k_encode<<<nodeGrid, 64, 0, stream>>>(ff, 1, P(17), P(18), P(19), P(20), P(21), P(22), P(23), P(24), P(25), fe);
  k_encode<<<nodeGrid, 64, 0, stream>>>(af, 5, P(26), P(27), P(28), P(29), P(30), P(31), P(32), P(33), P(34), ae);

  const int* src = ei;
  const int* dst = ei + E_;

  // impute (param base 35)
  auto impute = [&](const float* emb, const int* mask, float* outp) {
    hipMemsetAsync(nb, 0, (size_t)EMB * sizeof(float), stream);
    k_scatter<<<dim3(E_, B_), 64, 0, stream>>>(emb, src, dst, nb);
    k_impute<<<nodeGrid, 64, 0, stream>>>(nb, emb, mask, P(35), P(36), P(37), P(38), outp);
  };
  impute(pe, pm, pei);
  impute(fe, fm, fei);
  impute(ae, am, aei);

  // zero the three fusion accumulators (contiguous)
  hipMemsetAsync(pfu, 0, (size_t)3 * EMB * sizeof(float), stream);

  // cross attentions (param bases: p2f=39, p2a=49, f2p=59, f2a=69, a2p=79, a2f=89)
  auto xattn = [&](int base, const float* q, const float* kv, float* accb) {
    k_qkv<<<nodeGrid, 64, 0, stream>>>(q, kv, kv,
        P(base + 0), P(base + 1), P(base + 2), P(base + 3), P(base + 4), P(base + 5),
        Qh, Kh, Vh);
    k_attn<<<dim3(N_ / 16, NH_, B_), 32, 0, stream>>>(Qh, Kh, Vh, attb);
    k_projln<<<nodeGrid, 64, 0, stream>>>(attb, q,
        P(base + 6), P(base + 7), P(base + 8), P(base + 9), accb);
  };
  xattn(39, pei, fei, pfu);
  xattn(49, pei, aei, pfu);
  xattn(59, fei, pei, ffu);
  xattn(69, fei, aei, ffu);
  xattn(79, aei, pei, afu);
  xattn(89, aei, fei, afu);

  // fusion (modal_importance=99, fus=100..103)
  k_fuse<<<nodeGrid, 64, 0, stream>>>(pfu, ffu, afu, P(99), P(100), P(101), P(102), P(103), fused);

  float* outf = (float*)d_out;
  // node head (104..107) -> d_out[0 .. B*N)
  k_nodehead<<<nodeGrid, 64, 0, stream>>>(fused, P(104), P(105), P(106), P(107), outf);
  // graph head (108..111) -> d_out[B*N .. B*N+B)
  k_graphhead<<<dim3(B_), 64, 0, stream>>>(fused, P(108), P(109), P(110), P(111), outf + (size_t)B_ * N_);
}
